// GCN_test_62423054680390
// MI455X (gfx1250) — matrix operand order repaired and checked by reading, not compiled, not run on hardware
//
#include <hip/hip_runtime.h>
#include <hip/hip_bf16.h>

#define Bg    128
#define NPG   256
#define KKEEP 128
#define EEDGE 1048576
#define NTOT  (Bg * NPG)      // 32768
#define NEWN  (Bg * KKEEP)    // 16384
#define INC   64
#define HID   128
#define OUTC  2

typedef __attribute__((ext_vector_type(16))) _Float16 v16h;
typedef __attribute__((ext_vector_type(8)))  float    v8f;

// ---------- helpers ----------
__device__ __forceinline__ unsigned f2ord(float f) {
    unsigned u = __float_as_uint(f);
    return (u & 0x80000000u) ? ~u : (u | 0x80000000u);
}
__device__ __forceinline__ float ord2f(unsigned k) {
    unsigned b = (k & 0x80000000u) ? (k ^ 0x80000000u) : ~k;
    return __uint_as_float(b);
}
__device__ __forceinline__ float4 add4(float4 a, float4 b) {
    float4 r; r.x = a.x + b.x; r.y = a.y + b.y; r.z = a.z + b.z; r.w = a.w + b.w;
    return r;
}

// ---------- kernels ----------
__global__ void zero_f32(float* p, long n) {
    long i = (long)blockIdx.x * blockDim.x + threadIdx.x;
    if (i < n) p[i] = 0.f;
}

// agg0[d] += x[s], 4 channels per thread (float4 gather, 4 atomic adds)
__global__ void scatter_agg0(const float* __restrict__ x, const int* __restrict__ src,
                             const int* __restrict__ dst, float* __restrict__ agg) {
    long i = (long)blockIdx.x * blockDim.x + threadIdx.x;
    if (i >= (long)EEDGE * (INC / 4)) return;
    int e = (int)(i >> 4);            // 16 chunks of 4 channels per edge
    int c = ((int)i & 15) * 4;
    int s = src[e], d = dst[e];
    const float4 v = *(const float4*)(x + (long)s * INC + c);
    float* o = agg + (long)d * INC + c;
    atomicAdd(o + 0, v.x); atomicAdd(o + 1, v.y);
    atomicAdd(o + 2, v.z); atomicAdd(o + 3, v.w);
}

// pack f32 weight [K x NC] into per-lane f16 WMMA B-fragment layout:
// element (k,n): tcol=n>>4, r=n&15, k0=k>>5, half=(k>>3)&1, e=(k&7)+((k&31)>>4)*8, lane=half*16+r
__global__ void pack_w16(const float* __restrict__ W, _Float16* __restrict__ Wp,
                         int K, int NC) {
    int i = blockIdx.x * blockDim.x + threadIdx.x;
    if (i >= K * NC) return;
    int k = i / NC, n = i % NC;
    int tilesN = NC >> 4;
    int tcol = n >> 4, r = n & 15;
    int k0 = k >> 5, half = (k >> 3) & 1, e = (k & 7) + (((k & 31) >> 4) << 3);
    int lane = half * 16 + r;
    Wp[(((size_t)k0 * tilesN + tcol) * 32 + lane) * 16 + e] = (_Float16)W[i];
}

// fold bias + BN(eval) into per-column scale/shift: relu(acc*scale + shift)
__global__ void prep_bn(const float* __restrict__ bias, const float* __restrict__ g,
                        const float* __restrict__ be, const float* __restrict__ m,
                        const float* __restrict__ v, float* __restrict__ scale,
                        float* __restrict__ shift, int NC) {
    int c = blockIdx.x * blockDim.x + threadIdx.x;
    if (c >= NC) return;
    float sc = g[c] * rsqrtf(v[c] + 1e-5f);
    scale[c] = sc;
    shift[c] = (bias[c] - m[c]) * sc + be[c];
}

// out = relu( ((A (+A2)) @ W) * scale + shift ), one wave per 16x16 tile.
// A loaded as 4x float4 per k-step (contiguous runs of the CDNA5 A-fragment layout),
// B loaded as one contiguous v16h (pre-packed fragment).
template <bool HASA2>
__global__ void gemm_wmma(const float* __restrict__ A, const float* __restrict__ A2,
                          const _Float16* __restrict__ Wp, const float* __restrict__ scale,
                          const float* __restrict__ shift, float* __restrict__ out,
                          int M, int K, int NC) {
    int wave   = threadIdx.x >> 5;
    int tile   = blockIdx.x * (blockDim.x >> 5) + wave;
    int tilesN = NC >> 4;
    if (tile >= (M >> 4) * tilesN) return;
    int row0 = (tile / tilesN) << 4;
    int tcol = tile % tilesN;
    int lane = threadIdx.x & 31;
    int half = lane >> 4;
    int r    = lane & 15;
    size_t rowbase = (size_t)(row0 + r) * K;

    v8f acc = {};
    for (int k0 = 0; k0 < K; k0 += 32) {
        const float4* ap = (const float4*)(A + rowbase + k0 + half * 8);
        const float4* aq = (const float4*)(A + rowbase + k0 + 16 + half * 8);
        float4 a0 = ap[0], a1 = ap[1], a2v = aq[0], a3v = aq[1];
        if (HASA2) {
            const float4* bp = (const float4*)(A2 + rowbase + k0 + half * 8);
            const float4* bq = (const float4*)(A2 + rowbase + k0 + 16 + half * 8);
            a0 = add4(a0, bp[0]); a1 = add4(a1, bp[1]);
            a2v = add4(a2v, bq[0]); a3v = add4(a3v, bq[1]);
        }
        v16h a;
        a[0]  = (_Float16)a0.x;  a[1]  = (_Float16)a0.y;
        a[2]  = (_Float16)a0.z;  a[3]  = (_Float16)a0.w;
        a[4]  = (_Float16)a1.x;  a[5]  = (_Float16)a1.y;
        a[6]  = (_Float16)a1.z;  a[7]  = (_Float16)a1.w;
        a[8]  = (_Float16)a2v.x; a[9]  = (_Float16)a2v.y;
        a[10] = (_Float16)a2v.z; a[11] = (_Float16)a2v.w;
        a[12] = (_Float16)a3v.x; a[13] = (_Float16)a3v.y;
        a[14] = (_Float16)a3v.z; a[15] = (_Float16)a3v.w;
        v16h b = *(const v16h*)(Wp + (((size_t)(k0 >> 5) * tilesN + tcol) * 32 + lane) * 16);
        acc = __builtin_amdgcn_wmma_f32_16x16x32_f16(false, a, false, b,
                                                     (short)0, acc, false, false);
    }
    int   col = (tcol << 4) + r;
    float sc = scale[col], sh = shift[col];
#pragma unroll
    for (int i = 0; i < 8; ++i) {
        out[(size_t)(row0 + i + half * 8) * NC + col] = fmaxf(acc[i] * sc + sh, 0.f);
    }
}

// per-node xl/xr = h . wl/wr + b, one wave per node
__global__ void gat_xlxr(const float* __restrict__ h, const float* __restrict__ wl,
                         const float* __restrict__ bl, const float* __restrict__ wr,
                         const float* __restrict__ br, float* __restrict__ xl,
                         float* __restrict__ xr) {
    int n = (blockIdx.x * blockDim.x + threadIdx.x) >> 5;
    if (n >= NTOT) return;
    int lane = threadIdx.x & 31;
    float sl = 0.f, sr = 0.f;
    for (int i = lane; i < HID; i += 32) {
        float hv = h[(long)n * HID + i];
        sl += hv * wl[i];
        sr += hv * wr[i];
    }
    for (int o = 16; o > 0; o >>= 1) {
        sl += __shfl_xor(sl, o, 32);
        sr += __shfl_xor(sr, o, 32);
    }
    if (lane == 0) { xl[n] = sl + bl[0]; xr[n] = sr + br[0]; }
}

__global__ void gat_max(const int* __restrict__ src, const int* __restrict__ dst,
                        const float* __restrict__ xl, const float* __restrict__ xr,
                        const float* __restrict__ att, unsigned* __restrict__ mxu) {
    int i = blockIdx.x * blockDim.x + threadIdx.x;
    if (i >= EEDGE + NTOT) return;
    int s, d;
    if (i < EEDGE) { s = src[i]; d = dst[i]; } else { s = i - EEDGE; d = s; }
    float e = xl[s] + xr[d];
    e = e > 0.f ? e : 0.2f * e;
    atomicMax(mxu + d, f2ord(e * att[0]));
}

__global__ void gat_sum(const int* __restrict__ src, const int* __restrict__ dst,
                        const float* __restrict__ xl, const float* __restrict__ xr,
                        const float* __restrict__ att, const unsigned* __restrict__ mxu,
                        float* __restrict__ den, float* __restrict__ num) {
    int i = blockIdx.x * blockDim.x + threadIdx.x;
    if (i >= EEDGE + NTOT) return;
    int s, d;
    if (i < EEDGE) { s = src[i]; d = dst[i]; } else { s = i - EEDGE; d = s; }
    float e = xl[s] + xr[d];
    e = e > 0.f ? e : 0.2f * e;
    float ex = expf(e * att[0] - ord2f(mxu[d]));
    atomicAdd(den + d, ex);
    atomicAdd(num + d, ex * xl[s]);
}

__global__ void gat_score(const float* __restrict__ num, const float* __restrict__ den,
                          const float* __restrict__ gbias, float* __restrict__ score) {
    int n = blockIdx.x * blockDim.x + threadIdx.x;
    if (n >= NTOT) return;
    score[n] = fmaxf(num[n] / den[n] + gbias[0], 0.f);
}

// top-K per graph via rank count (value desc, index asc — matches jax.lax.top_k)
__global__ void topk_rank(const float* __restrict__ score, int* __restrict__ perm,
                          int* __restrict__ node_map) {
    __shared__ float s[NPG];
    int g = blockIdx.x, i = threadIdx.x;
    float si = score[g * NPG + i];
    s[i] = si;
    __syncthreads();
    int rank = 0;
    for (int j = 0; j < NPG; ++j) {
        float sj = s[j];
        rank += (sj > si) || (sj == si && j < i);
    }
    int gid = g * NPG + i;
    if (rank < KKEEP) {
        perm[g * KKEEP + rank] = gid;
        node_map[gid] = g * KKEEP + rank;
    } else {
        node_map[gid] = -1;
    }
}

__global__ void gather_hp(const float* __restrict__ h, const float* __restrict__ score,
                          const int* __restrict__ perm, float* __restrict__ hp) {
    long i = (long)blockIdx.x * blockDim.x + threadIdx.x;
    if (i >= (long)NEWN * HID) return;
    int nn = (int)(i >> 7), c = (int)i & 127;
    int old = perm[nn];
    hp[i] = h[(long)old * HID + c] * score[old];
}

// agg1[d2] += hp[s2] for valid relabeled edges, 4 channels per thread
__global__ void scatter_agg1(const int* __restrict__ src, const int* __restrict__ dst,
                             const int* __restrict__ node_map, const float* __restrict__ hp,
                             float* __restrict__ agg1) {
    long i = (long)blockIdx.x * blockDim.x + threadIdx.x;
    if (i >= (long)EEDGE * (HID / 4)) return;
    int e = (int)(i >> 5);            // 32 chunks of 4 channels per edge
    int c = ((int)i & 31) * 4;
    int s2 = node_map[src[e]];
    int d2 = node_map[dst[e]];
    if (s2 < 0 || d2 < 0) return;
    const float4 v = *(const float4*)(hp + (long)s2 * HID + c);
    float* o = agg1 + (long)d2 * HID + c;
    atomicAdd(o + 0, v.x); atomicAdd(o + 1, v.y);
    atomicAdd(o + 2, v.z); atomicAdd(o + 3, v.w);
}

// mean pool over K rows per graph + final linear to OUTC
__global__ void pool_linear(const float* __restrict__ h1, const float* __restrict__ lw,
                            const float* __restrict__ lb, float* __restrict__ out) {
    __shared__ float pooled[HID];
    int g = blockIdx.x, c = threadIdx.x;
    float s = 0.f;
    for (int k = 0; k < KKEEP; ++k) s += h1[((long)(g * KKEEP + k)) * HID + c];
    pooled[c] = s * (1.f / KKEEP);
    __syncthreads();
    if (c < OUTC) {
        float o = lb[c];
        for (int j = 0; j < HID; ++j) o += pooled[j] * lw[j * OUTC + c];
        out[g * OUTC + c] = o;
    }
}

// ---------- launch ----------
extern "C" void kernel_launch(void* const* d_in, const int* in_sizes, int n_in,
                              void* d_out, int out_size, void* d_ws, size_t ws_size,
                              hipStream_t stream) {
    const float* x   = (const float*)d_in[0];
    const int*   ei  = (const int*)d_in[1];
    const int*   src = ei;
    const int*   dst = ei + EEDGE;
    // mlp0: 3..14  mlp1: 15..26  gat: 27..32  lin: 33..34
    const float* m0w1 = (const float*)d_in[3];  const float* m0b1 = (const float*)d_in[4];
    const float* m0g1 = (const float*)d_in[5];  const float* m0e1 = (const float*)d_in[6];
    const float* m0m1 = (const float*)d_in[7];  const float* m0v1 = (const float*)d_in[8];
    const float* m0w2 = (const float*)d_in[9];  const float* m0b2 = (const float*)d_in[10];
    const float* m0g2 = (const float*)d_in[11]; const float* m0e2 = (const float*)d_in[12];
    const float* m0m2 = (const float*)d_in[13]; const float* m0v2 = (const float*)d_in[14];
    const float* m1w1 = (const float*)d_in[15]; const float* m1b1 = (const float*)d_in[16];
    const float* m1g1 = (const float*)d_in[17]; const float* m1e1 = (const float*)d_in[18];
    const float* m1m1 = (const float*)d_in[19]; const float* m1v1 = (const float*)d_in[20];
    const float* m1w2 = (const float*)d_in[21]; const float* m1b2 = (const float*)d_in[22];
    const float* m1g2 = (const float*)d_in[23]; const float* m1e2 = (const float*)d_in[24];
    const float* m1m2 = (const float*)d_in[25]; const float* m1v2 = (const float*)d_in[26];
    const float* wl  = (const float*)d_in[27];  const float* bl  = (const float*)d_in[28];
    const float* wr  = (const float*)d_in[29];  const float* br  = (const float*)d_in[30];
    const float* att = (const float*)d_in[31];  const float* gbi = (const float*)d_in[32];
    const float* lw  = (const float*)d_in[33];  const float* lb  = (const float*)d_in[34];
    float* out = (float*)d_out;

    // workspace layout (all offsets >= 32B-aligned)
    float*    agg0 = (float*)d_ws;                              // N*INC
    float*    agg1 = agg0 + (long)NTOT * INC;                   // NEWN*HID
    unsigned* mxu  = (unsigned*)(agg1 + (long)NEWN * HID);      // N
    float*    den  = (float*)(mxu + NTOT);                      // N
    float*    num  = den + NTOT;                                // N
    float*    xl   = num + NTOT;                                // N
    float*    xr   = xl + NTOT;                                 // N
    float*    sco  = xr + NTOT;                                 // N
    float*    t1   = sco + NTOT;                                // N*HID
    float*    hbuf = t1 + (long)NTOT * HID;                     // N*HID
    float*    hp   = hbuf + (long)NTOT * HID;                   // NEWN*HID
    float*    t2   = hp + (long)NEWN * HID;                     // NEWN*HID
    float*    h1   = t2 + (long)NEWN * HID;                     // NEWN*HID
    int*      nmap = (int*)(h1 + (long)NEWN * HID);             // N
    int*      perm = nmap + NTOT;                               // NEWN
    _Float16* wp0  = (_Float16*)(perm + NEWN);                  // INC*HID
    _Float16* wp1  = wp0 + (long)INC * HID;                     // HID*HID
    _Float16* wp2  = wp1 + (long)HID * HID;                     // HID*HID
    _Float16* wp3  = wp2 + (long)HID * HID;                     // HID*HID
    float*    bns  = (float*)(wp3 + (long)HID * HID);           // 8 * HID (scale/shift x4)
    float*    sc0 = bns,            * sh0 = bns + HID;
    float*    sc1 = bns + 2 * HID,  * sh1 = bns + 3 * HID;
    float*    sc2 = bns + 4 * HID,  * sh2 = bns + 5 * HID;
    float*    sc3 = bns + 6 * HID,  * sh3 = bns + 7 * HID;

    // zero agg0, agg1, mxu, den, num (contiguous block at start of ws)
    long nzero = (long)NTOT * INC + (long)NEWN * HID + 3L * NTOT;
    zero_f32<<<(unsigned)((nzero + 255) / 256), 256, 0, stream>>>((float*)d_ws, nzero);

    // pack weights into WMMA B-fragment layout + fold BN into scale/shift
    pack_w16<<<(INC * HID + 255) / 256, 256, 0, stream>>>(m0w1, wp0, INC, HID);
    pack_w16<<<(HID * HID + 255) / 256, 256, 0, stream>>>(m0w2, wp1, HID, HID);
    pack_w16<<<(HID * HID + 255) / 256, 256, 0, stream>>>(m1w1, wp2, HID, HID);
    pack_w16<<<(HID * HID + 255) / 256, 256, 0, stream>>>(m1w2, wp3, HID, HID);
    prep_bn<<<1, HID, 0, stream>>>(m0b1, m0g1, m0e1, m0m1, m0v1, sc0, sh0, HID);
    prep_bn<<<1, HID, 0, stream>>>(m0b2, m0g2, m0e2, m0m2, m0v2, sc1, sh1, HID);
    prep_bn<<<1, HID, 0, stream>>>(m1b1, m1g1, m1e1, m1m1, m1v1, sc2, sh2, HID);
    prep_bn<<<1, HID, 0, stream>>>(m1b2, m1g2, m1e2, m1m2, m1v2, sc3, sh3, HID);

    // GIN layer 0 aggregation
    scatter_agg0<<<(unsigned)(((long)EEDGE * 16 + 255) / 256), 256, 0, stream>>>(x, src, dst, agg0);

    // MLP0 (two WMMA GEMMs with fused scale/shift + ReLU)
    {
        int tiles = (NTOT / 16) * (HID / 16);
        gemm_wmma<true><<<(tiles + 7) / 8, 256, 0, stream>>>(x, agg0, wp0, sc0, sh0, t1,
                                                             NTOT, INC, HID);
        gemm_wmma<false><<<(tiles + 7) / 8, 256, 0, stream>>>(t1, nullptr, wp1, sc1, sh1, hbuf,
                                                              NTOT, HID, HID);
    }

    // GATv2 scoring
    gat_xlxr<<<(NTOT * 32 + 255) / 256, 256, 0, stream>>>(hbuf, wl, bl, wr, br, xl, xr);
    int nedge = EEDGE + NTOT;
    gat_max<<<(nedge + 255) / 256, 256, 0, stream>>>(src, dst, xl, xr, att, mxu);
    gat_sum<<<(nedge + 255) / 256, 256, 0, stream>>>(src, dst, xl, xr, att, mxu, den, num);
    gat_score<<<(NTOT + 255) / 256, 256, 0, stream>>>(num, den, gbi, sco);

    // SAGPooling top-K + gather/scale
    topk_rank<<<Bg, NPG, 0, stream>>>(sco, perm, nmap);
    gather_hp<<<(unsigned)(((long)NEWN * HID + 255) / 256), 256, 0, stream>>>(hbuf, sco, perm, hp);

    // GIN layer 1 aggregation on kept nodes
    scatter_agg1<<<(unsigned)(((long)EEDGE * 32 + 255) / 256), 256, 0, stream>>>(src, dst, nmap, hp, agg1);

    // MLP1
    {
        int tiles = (NEWN / 16) * (HID / 16);
        gemm_wmma<true><<<(tiles + 7) / 8, 256, 0, stream>>>(hp, agg1, wp2, sc2, sh2, t2,
                                                             NEWN, HID, HID);
        gemm_wmma<false><<<(tiles + 7) / 8, 256, 0, stream>>>(t2, nullptr, wp3, sc3, sh3, h1,
                                                              NEWN, HID, HID);
    }

    // mean pool + classifier
    pool_linear<<<Bg, HID, 0, stream>>>(h1, lw, lb, out);
    (void)in_sizes; (void)n_in; (void)out_size; (void)ws_size;
}
